// SDConv_85134841741763
// MI455X (gfx1250) — matrix operand order repaired and checked by reading, not compiled, not run on hardware
//
#include <hip/hip_runtime.h>

// Problem constants (fixed by the reference)
#define HEADS   16
#define HS      64
#define KTAPS   9
#define PADW    4
#define SEQ     4096
#define BATCH   4
#define HIDDEN  (HEADS * HS)        // 1024

// Tiling
#define TILE       128              // seq rows per block
#define HALO_ROWS  (TILE + 2*PADW)  // 136
#define QROW       72               // bf16 row stride: 144B, 16B-aligned, conflict-free
#define BROW       584              // bf16 row stride: 1168B, 16B-aligned
#define KC         576              // contraction length = HS * KTAPS
#define NCHUNK     (KC / 32)        // 18 WMMA k-steps

typedef __attribute__((ext_vector_type(16))) __bf16 v16bf;
typedef __attribute__((ext_vector_type(8)))  __bf16 v8bf;
typedef __attribute__((ext_vector_type(8)))  float  v8f;

__device__ __forceinline__ __bf16 f2bf(float f) { return (__bf16)f; }

__global__ __launch_bounds__(256, 2)
void sdconv_wmma_kernel(const float* __restrict__ q,
                        const float* __restrict__ w,
                        float* __restrict__ out)
{
    // A staging: query tile as bf16, contraction order kk = t*64 + d, so the
    // WMMA A-rows are overlapping windows of this flat tile.
    __shared__ __align__(16) __bf16 qa[HALO_ROWS * QROW];
    // B staging: per-head predictor weights, B[n=tap][kk], zero-padded to N=16.
    __shared__ __align__(16) __bf16 bw[16 * BROW];
    // Predictor logits (16x16 per wave; only 9 cols valid).
    __shared__ __align__(16) float  lg[TILE * 16];

    const int tid = threadIdx.x;
    const int s0  = blockIdx.x * TILE;
    const int h   = blockIdx.y;
    const int bb  = blockIdx.z;

    const float* qb = q   + (size_t)bb * SEQ * HIDDEN + h * HS;
    float*       ob = out + (size_t)bb * SEQ * HIDDEN + h * HS;

    // ---- zero B region (covers tap rows 9..15 and stride padding) ----
    for (int i = tid; i < (16 * BROW) / 2; i += 256)
        reinterpret_cast<unsigned int*>(bw)[i] = 0u;

    // ---- fill A: 136 halo rows x 64 dims, fp32 -> bf16 ----
    for (int idx = tid; idx < HALO_ROWS * (HS / 4); idx += 256) {
        int row = idx >> 4;
        int c4  = (idx & 15) << 2;
        int g   = s0 - PADW + row;
        float4 v = make_float4(0.f, 0.f, 0.f, 0.f);
        if (g >= 0 && g < SEQ)
            v = *reinterpret_cast<const float4*>(qb + (size_t)g * HIDDEN + c4);
        union { __bf16 b[4]; uint2 u; } pk;
        pk.b[0] = f2bf(v.x); pk.b[1] = f2bf(v.y);
        pk.b[2] = f2bf(v.z); pk.b[3] = f2bf(v.w);
        *reinterpret_cast<uint2*>(&qa[row * QROW + c4]) = pk.u;
    }

    // Warm GL2 for the stage-2 fp32 re-read of this tile.
    __builtin_prefetch(qb + (size_t)s0 * HIDDEN + (tid << 4), 0, 0);

    __syncthreads();   // zeroing of bw complete before scatter-fill

    // ---- fill B: W[h, k, d, t] -> bw[k][t*64 + d] as bf16 ----
    for (int i = tid; i < KTAPS * KC; i += 256) {
        int k  = i / KC;
        int kk = i - k * KC;
        int t  = kk >> 6;
        int d  = kk & 63;
        float wv = w[((size_t)((h * KTAPS + k) * HS + d)) * KTAPS + t];
        bw[k * BROW + kk] = f2bf(wv);
    }
    __syncthreads();

    // ---- Stage 1: predictor GEMM via v_wmma_f32_16x16x32_bf16 ----
    // Wave 'wave' computes logits for 16 seq rows [m0, m0+16).
    const int wave = tid >> 5;
    const int lane = tid & 31;
    const int half = lane >> 4;   // ISA bf16 A/B layout: lane halves split K
    const int lm   = lane & 15;
    const int m0   = wave << 4;

    v8f acc = {};
    const __bf16* arow = &qa[(m0 + lm) * QROW + half * 8];
    const __bf16* brow = &bw[lm * BROW + half * 8];
    #pragma unroll
    for (int c = 0; c < NCHUNK; ++c) {
        // A chunk c covers kk = c*32..c*32+31  ->  tap t = c/2, dim d = (c&1)*32 + ...
        const __bf16* ap = arow + (c >> 1) * QROW + (c & 1) * 32;
        v8bf a_lo = *reinterpret_cast<const v8bf*>(ap);        // K = c*32 + half*8 + 0..7
        v8bf a_hi = *reinterpret_cast<const v8bf*>(ap + 16);   // K = c*32 + 16 + half*8 + 0..7
        v16bf av;
        #pragma unroll
        for (int i = 0; i < 8; ++i) { av[i] = a_lo[i]; av[i + 8] = a_hi[i]; }

        const __bf16* bp = brow + c * 32;                      // K = c*32 + half*8 + 0..7, c*32 + 16 + half*8 + 0..7
        v8bf b_lo = *reinterpret_cast<const v8bf*>(bp);
        v8bf b_hi = *reinterpret_cast<const v8bf*>(bp + 16);
        v16bf bv;
        #pragma unroll
        for (int i = 0; i < 8; ++i) { bv[i] = b_lo[i]; bv[i + 8] = b_hi[i]; }

        acc = __builtin_amdgcn_wmma_f32_16x16x32_bf16(
                  false, av, false, bv, (short)0, acc, false, false);
    }

    // C layout: VGPR r, lane -> row m0 + half*8 + r, col lm
    #pragma unroll
    for (int rr = 0; rr < 8; ++rr)
        lg[(m0 + half * 8 + rr) * 16 + lm] = acc[rr];
    __syncthreads();

    // ---- Stage 2: softmax over taps + weighted window sum (fp32) ----
    // 16 lanes span one row's 64 dims (float4 each) -> coalesced 256B rows.
    for (int pass = 0; pass < 8; ++pass) {
        int r  = pass * 16 + (tid >> 4);   // 0..127
        int d4 = (tid & 15) << 2;          // 0..60
        const float* lrow = &lg[r * 16];

        float mx = lrow[0];
        #pragma unroll
        for (int k = 1; k < KTAPS; ++k) mx = fmaxf(mx, lrow[k]);
        float wt[KTAPS];
        float sum = 0.f;
        #pragma unroll
        for (int k = 0; k < KTAPS; ++k) { wt[k] = __expf(lrow[k] - mx); sum += wt[k]; }
        float inv = 1.0f / sum;

        float a0 = 0.f, a1 = 0.f, a2 = 0.f, a3 = 0.f;
        #pragma unroll
        for (int k = 0; k < KTAPS; ++k) {
            int g = s0 + r + k - PADW;
            if (g >= 0 && g < SEQ) {
                float4 v = *reinterpret_cast<const float4*>(qb + (size_t)g * HIDDEN + d4);
                float ww = wt[k] * inv;
                a0 = fmaf(ww, v.x, a0);
                a1 = fmaf(ww, v.y, a1);
                a2 = fmaf(ww, v.z, a2);
                a3 = fmaf(ww, v.w, a3);
            }
        }
        *reinterpret_cast<float4*>(ob + (size_t)(s0 + r) * HIDDEN + d4) =
            make_float4(a0, a1, a2, a3);
    }
}

extern "C" void kernel_launch(void* const* d_in, const int* in_sizes, int n_in,
                              void* d_out, int out_size, void* d_ws, size_t ws_size,
                              hipStream_t stream) {
    (void)in_sizes; (void)n_in; (void)out_size; (void)d_ws; (void)ws_size;
    const float* q = (const float*)d_in[0];   // (4, 4096, 1024) fp32
    const float* w = (const float*)d_in[1];   // (144, 64, 9)    fp32
    float* out = (float*)d_out;               // (4, 4096, 1024) fp32

    dim3 grid(SEQ / TILE, HEADS, BATCH);      // 32 x 16 x 4 = 2048 blocks
    sdconv_wmma_kernel<<<grid, 256, 0, stream>>>(q, w, out);
}
